// DMFA_26345329393684
// MI455X (gfx1250) — matrix operand order repaired and checked by reading, not compiled, not run on hardware
//
#include <hip/hip_runtime.h>
#include <math.h>

// ---------------- problem constants ----------------
#define BATCH 8
#define DIMC  96
#define HH    128
#define WW    128
#define HWN   16384          // H*W
#define NPIX  (BATCH * HWN)  // 131072
#define NHEADS 4
#define CHD   24             // DIMC / NHEADS
#define HID   255
#define HID2  510

typedef _Float16 half16 __attribute__((ext_vector_type(16)));
typedef _Float16 half8  __attribute__((ext_vector_type(8)));
typedef float    float8 __attribute__((ext_vector_type(8)));

// ---------------- helpers ----------------
__device__ __forceinline__ half16 dmfa_load16(const float* p) {
    float buf[16];
    *(float4*)(buf + 0)  = ((const float4*)p)[0];
    *(float4*)(buf + 4)  = ((const float4*)p)[1];
    *(float4*)(buf + 8)  = ((const float4*)p)[2];
    *(float4*)(buf + 12) = ((const float4*)p)[3];
    half16 r;
#pragma unroll
    for (int i = 0; i < 16; ++i) r[i] = (_Float16)buf[i];
    return r;
}
__device__ __forceinline__ half16 dmfa_load16(const _Float16* p) {
    return *(const half16*)p;
}

// ---------------- weight pad+convert (f32 -> f16, zero pad to M%16, K%32) ----
__global__ void dmfa_padw(const float* __restrict__ src, _Float16* __restrict__ dst,
                          int OC, int K, int Mp, int Kp) {
    int idx = blockIdx.x * blockDim.x + threadIdx.x;
    if (idx >= Mp * Kp) return;
    int m = idx / Kp, k = idx % Kp;
    dst[idx] = (m < OC && k < K) ? (_Float16)src[m * K + k] : (_Float16)0.0f;
}

// ---------------- bias-free LayerNorm over channels (C=96) ------------------
__global__ void dmfa_ln(const float* __restrict__ x, const float* __restrict__ w,
                        float* __restrict__ y) {
    int gid = blockIdx.x * blockDim.x + threadIdx.x;
    if (gid >= NPIX) return;
    int b = gid >> 14, hw = gid & (HWN - 1);
    const float* px = x + (size_t)b * DIMC * HWN + hw;
    float s = 0.f, ss = 0.f;
#pragma unroll 4
    for (int c = 0; c < DIMC; ++c) { float v = px[(size_t)c * HWN]; s += v; ss += v * v; }
    float mean = s * (1.0f / DIMC);
    float var  = ss * (1.0f / DIMC) - mean * mean;
    float r = rsqrtf(var + 1e-5f);
    float* py = y + (size_t)b * DIMC * HWN + hw;
#pragma unroll 4
    for (int c = 0; c < DIMC; ++c) py[(size_t)c * HWN] = px[(size_t)c * HWN] * r * w[c];
}

// ---------------- conv1x1 as WMMA GEMM: Y[oc,p] = sum_k Wh[oc,k]*X[k,p] ------
// Wh: padded f16 [Mp, Kp] row-major. X,Y: [B, C, HW] layout.
// One wave = 16(oc) x 32(pixels): two 16x16 WMMA tiles sharing the A (weight) tile.
// KGUARD: mask B loads when K < Kp       (only w_fo: K=255, Kp=256)
// MGUARD: mask stores when OC % 16 != 0  (only w_fi: OC=510)
// EPI: 0 = plain, 1 = + R1 + R2 (out1 = x+y+proj), 2 = + R1 (final residual add)
template <typename TIn, typename TOut, int EPI, bool KGUARD, bool MGUARD>
__global__ void dmfa_gemm16(const _Float16* __restrict__ Wh,
                            const TIn* __restrict__ X, TOut* __restrict__ Y,
                            const float* __restrict__ R1, const float* __restrict__ R2,
                            int OC, int K, int Kp) {
    const int lane  = threadIdx.x & 31;
    const int wave  = threadIdx.x >> 5;
    const int ntile = blockIdx.x * 4 + wave;   // 32-pixel tile id
    const int p0    = ntile << 5;
    const int b     = p0 >> 14;                // 32-pixel tiles never cross batch
    const int hw0   = p0 & (HWN - 1);
    const int m0    = blockIdx.y << 4;
    const TIn* xb   = X + (size_t)b * K * HWN + hw0;
    const int Kt    = Kp >> 5;
    // A layout (16-bit 16x32): lanes 0-15 hold K {0..7,16..23}, lanes 16-31 {8..15,24..31}
    const _Float16* wrow =
        Wh + (size_t)(m0 + (lane & 15)) * Kp + ((lane < 16) ? 0 : 8);
    float8 acc0 = {};
    float8 acc1 = {};
    for (int kt = 0; kt < Kt; ++kt) {
        half8 lo = *(const half8*)(wrow + kt * 32);
        half8 hi = *(const half8*)(wrow + kt * 32 + 16);
        half16 a;
#pragma unroll
        for (int i = 0; i < 8; ++i) { a[i] = lo[i]; a[i + 8] = hi[i]; }
        // B layout (32x16): lane%16 = N (pixel), element = K in the same order as A
        const int bh = lane >> 4, bn = lane & 15;
        half16 bv0, bv1;
#pragma unroll
        for (int s = 0; s < 16; ++s) {
            int k = (kt << 5) + ((s < 8) ? (8 * bh + s) : (16 + 8 * bh + (s - 8)));
            const TIn* brow = xb + (size_t)k * HWN + bn;
            bool ok = !KGUARD || k < K;
            bv0[s] = ok ? (_Float16)brow[0] : (_Float16)0.0f;
            bv1[s] = ok ? (_Float16)brow[16] : (_Float16)0.0f;
        }
        acc0 = __builtin_amdgcn_wmma_f32_16x16x32_f16(
            false, a, false, bv0, (short)0, acc0, false, false);
        acc1 = __builtin_amdgcn_wmma_f32_16x16x32_f16(
            false, a, false, bv1, (short)0, acc1, false, false);
    }
    // D layout: lane = N (pixel), VGPR r -> M = r + 8*(lane/16)
    const int col = lane & 15, rg = lane >> 4;
    const size_t outb = (size_t)b * OC * HWN + (size_t)(hw0 + col);
#pragma unroll
    for (int r = 0; r < 8; ++r) {
        int oc = m0 + r + rg * 8;
        if (!MGUARD || oc < OC) {
            size_t a0 = outb + (size_t)oc * HWN;
            size_t a1 = a0 + 16;
            float v0 = acc0[r];
            float v1 = acc1[r];
            if (EPI == 1) { v0 += R1[a0] + R2[a0]; v1 += R1[a1] + R2[a1]; }
            if (EPI == 2) { v0 += R1[a0]; v1 += R1[a1]; }
            Y[a0] = (TOut)v0;
            Y[a1] = (TOut)v1;
        }
    }
}

// ---------------- depthwise 3x3, SAME zero pad ------------------------------
template <typename T>
__global__ void dmfa_dwconv(const T* __restrict__ x, const float* __restrict__ w,
                            T* __restrict__ y, int Cn, int total) {
    int gid = blockIdx.x * blockDim.x + threadIdx.x;
    if (gid >= total) return;
    int hw = gid & (HWN - 1);
    int c = (gid >> 14) % Cn;
    int i = hw >> 7, j = hw & (WW - 1);
    const float* wc = w + c * 9;
    float acc = 0.f;
#pragma unroll
    for (int di = 0; di < 3; ++di) {
        int ii = i + di - 1;
        if (ii < 0 || ii >= HH) continue;
#pragma unroll
        for (int dj = 0; dj < 3; ++dj) {
            int jj = j + dj - 1;
            if (jj < 0 || jj >= WW) continue;
            acc += wc[di * 3 + dj] * (float)x[(size_t)gid + (di - 1) * WW + (dj - 1)];
        }
    }
    y[gid] = (T)acc;
}

// ---------------- row-wise L2 normalize over HW (in place) ------------------
// row r -> channel (r%CS) of batch (r/CS) in a tensor with CT channels/batch
__global__ void dmfa_l2norm(float* __restrict__ p, int CS, int CT) {
    __shared__ float red[256];
    int row = blockIdx.x;
    float* pr = p + ((size_t)(row / CS) * CT + (row % CS)) * HWN;
    int tid = threadIdx.x;
    float ss = 0.f;
    for (int i = tid; i < HWN; i += 256) { float v = pr[i]; ss += v * v; }
    red[tid] = ss;
    __syncthreads();
    for (int s = 128; s > 0; s >>= 1) {
        if (tid < s) red[tid] += red[tid + s];
        __syncthreads();
    }
    float inv = 1.0f / fmaxf(sqrtf(red[0]), 1e-12f);
    for (int i = tid; i < HWN; i += 256) pr[i] *= inv;
}

// ---------------- attn = softmax(temp * q k^T) per (b, head) ---------------
__global__ void dmfa_attn(const float* __restrict__ q, const float* __restrict__ kv,
                          const float* __restrict__ temp, float* __restrict__ attn) {
    __shared__ float qs[CHD][132];
    __shared__ float ks[CHD][132];
    __shared__ float S[CHD][CHD];
    int bh = blockIdx.x, b = bh >> 2, h = bh & 3;
    int tid = threadIdx.x;
    const float* qb = q  + ((size_t)b * DIMC + h * CHD) * HWN;
    const float* kb = kv + ((size_t)b * (2 * DIMC) + h * CHD) * HWN;  // k = first 96 ch
    float a0 = 0.f, a1 = 0.f, a2 = 0.f;
    int i0 = tid, c0 = i0 / CHD, d0 = i0 % CHD;
    int i1 = tid + 256, c1 = i1 / CHD, d1 = i1 % CHD;
    int i2 = tid + 512, c2 = i2 / CHD, d2 = i2 % CHD;
    for (int n0 = 0; n0 < HWN; n0 += 128) {
        for (int i = tid; i < CHD * 128; i += 256) {
            int c = i >> 7, t = i & 127;
            qs[c][t] = qb[(size_t)c * HWN + n0 + t];
            ks[c][t] = kb[(size_t)c * HWN + n0 + t];
        }
        __syncthreads();
        for (int t = 0; t < 128; ++t) {
            a0 += qs[c0][t] * ks[d0][t];
            a1 += qs[c1][t] * ks[d1][t];
            if (i2 < CHD * CHD) a2 += qs[c2][t] * ks[d2][t];
        }
        __syncthreads();
    }
    float T = temp[h];
    S[c0][d0] = a0 * T;
    S[c1][d1] = a1 * T;
    if (i2 < CHD * CHD) S[c2][d2] = a2 * T;
    __syncthreads();
    if (tid < CHD) {
        float m = -1e30f;
        for (int d = 0; d < CHD; ++d) m = fmaxf(m, S[tid][d]);
        float sum = 0.f;
        float e[CHD];
        for (int d = 0; d < CHD; ++d) { e[d] = __expf(S[tid][d] - m); sum += e[d]; }
        float inv = 1.0f / sum;
        float* out = attn + ((size_t)bh * CHD + tid) * CHD;
        for (int d = 0; d < CHD; ++d) out[d] = e[d] * inv;
    }
}

// ---------------- out[c,n] = sum_d attn[c,d] * v[d,n] ----------------------
__global__ void dmfa_av(const float* __restrict__ attn, const float* __restrict__ kv,
                        float* __restrict__ out) {
    __shared__ float a[CHD];
    int tid = threadIdx.x;
    int cg = blockIdx.y, b = cg / DIMC, c = cg % DIMC;
    int h = c / CHD, cl = c % CHD;
    if (tid < CHD) a[tid] = attn[(((size_t)(b * NHEADS + h)) * CHD + cl) * CHD + tid];
    __syncthreads();
    int n = blockIdx.x * 256 + tid;
    const float* vb = kv + ((size_t)b * (2 * DIMC) + DIMC + h * CHD) * HWN + n;  // v = ch 96..191
    float acc = 0.f;
#pragma unroll
    for (int d = 0; d < CHD; ++d) acc += a[d] * vb[(size_t)d * HWN];
    out[((size_t)b * DIMC + c) * HWN + n] = acc;
}

// ---------------- g = gelu_exact(t1) * t2 ----------------------------------
__global__ void dmfa_gelugate(const _Float16* __restrict__ td, _Float16* __restrict__ g,
                              int total) {
    int gid = blockIdx.x * blockDim.x + threadIdx.x;
    if (gid >= total) return;
    int per = HID * HWN;
    int b = gid / per, r = gid % per;
    int c = r >> 14, hw = r & (HWN - 1);
    size_t i1 = ((size_t)b * HID2 + c) * HWN + hw;
    float x1 = (float)td[i1];
    float x2 = (float)td[i1 + (size_t)HID * HWN];
    float ge = 0.5f * x1 * (1.0f + erff(x1 * 0.70710678118654752f));
    g[gid] = (_Float16)(ge * x2);
}

// ---------------- launch ----------------------------------------------------
extern "C" void kernel_launch(void* const* d_in, const int* in_sizes, int n_in,
                              void* d_out, int out_size, void* d_ws, size_t ws_size,
                              hipStream_t stream) {
    const float* x       = (const float*)d_in[0];
    const float* y       = (const float*)d_in[1];
    const float* w_norm1 = (const float*)d_in[2];
    const float* w_norm2 = (const float*)d_in[3];
    const float* temp    = (const float*)d_in[4];
    const float* w_kv    = (const float*)d_in[5];
    const float* w_kv_dw = (const float*)d_in[6];
    const float* w_q     = (const float*)d_in[7];
    const float* w_q_dw  = (const float*)d_in[8];
    const float* w_proj  = (const float*)d_in[9];
    const float* w_fi    = (const float*)d_in[10];
    const float* w_fdw   = (const float*)d_in[11];
    const float* w_fo    = (const float*)d_in[12];
    float* out = (float*)d_out;
    char* ws = (char*)d_ws;

    const size_t SZ_C  = (size_t)BATCH * DIMC * HWN;  // 12.58M elems
    const size_t SZ_C2 = 2 * SZ_C;

    // f16 padded weight stash (~220 KB)
    size_t off = 0;
    _Float16* Wkvh = (_Float16*)(ws + off); off += (size_t)192 * 96 * 2;
    _Float16* Wqh  = (_Float16*)(ws + off); off += (size_t)96 * 96 * 2;
    _Float16* Wph  = (_Float16*)(ws + off); off += (size_t)96 * 96 * 2;
    _Float16* Wfih = (_Float16*)(ws + off); off += (size_t)512 * 96 * 2;
    _Float16* Wfoh = (_Float16*)(ws + off); off += (size_t)96 * 256 * 2;
    off = (off + 255) & ~(size_t)255;
    // activation regions (with aliasing for dead buffers)
    size_t off_xn = off;                float* xn   = (float*)(ws + off); off += SZ_C * 4;
    size_t off_yn = off;                float* yn   = (float*)(ws + off); off += SZ_C * 4;
    size_t off_kv = off;                float* kvr  = (float*)(ws + off); off += SZ_C2 * 4;
                                        float* qr   = (float*)(ws + off); off += SZ_C * 4;
    size_t off_kvd = off;               float* kvd  = (float*)(ws + off); off += SZ_C2 * 4;
                                        float* qd   = (float*)(ws + off); off += SZ_C * 4;
                                        float* attn = (float*)(ws + off);
    off += (size_t)BATCH * NHEADS * CHD * CHD * 4;
                                        float* out1 = (float*)(ws + off); off += SZ_C * 4;
    float*     attnout = (float*)(ws + off_xn);       // xn dead after kv GEMM
    float*     o1n     = (float*)(ws + off_yn);       // yn dead after q GEMM
    _Float16*  traw    = (_Float16*)(ws + off_kv);    // 133.7MB <= kv+q region (151MB)
    _Float16*  td      = (_Float16*)(ws + off_kvd);   // 133.7MB <= kvd+qd region
    _Float16*  g       = (_Float16*)(ws + off_xn);    // 66.8MB <= xn+yn region

    // 1) pad/convert weights to f16
    dmfa_padw<<<(192 * 96 + 255) / 256, 256, 0, stream>>>(w_kv, Wkvh, 192, 96, 192, 96);
    dmfa_padw<<<(96 * 96 + 255) / 256, 256, 0, stream>>>(w_q, Wqh, 96, 96, 96, 96);
    dmfa_padw<<<(96 * 96 + 255) / 256, 256, 0, stream>>>(w_proj, Wph, 96, 96, 96, 96);
    dmfa_padw<<<(512 * 96 + 255) / 256, 256, 0, stream>>>(w_fi, Wfih, 510, 96, 512, 96);
    dmfa_padw<<<(96 * 256 + 255) / 256, 256, 0, stream>>>(w_fo, Wfoh, 96, 255, 96, 256);

    // 2) bias-free LN
    dmfa_ln<<<NPIX / 256, 256, 0, stream>>>(x, w_norm1, xn);
    dmfa_ln<<<NPIX / 256, 256, 0, stream>>>(y, w_norm2, yn);

    const int gx = NPIX / 32 / 4;  // 1024 blocks x 4 waves = 4096 32-pixel tiles
    // 3) kv = W_kv @ xn   (192 x 131072, K=96)
    dmfa_gemm16<float, float, 0, false, false><<<dim3(gx, 12), 128, 0, stream>>>(
        Wkvh, xn, kvr, nullptr, nullptr, 192, 96, 96);
    // 4) depthwise 3x3 on kv
    dmfa_dwconv<float><<<((int)SZ_C2 + 255) / 256, 256, 0, stream>>>(
        kvr, w_kv_dw, kvd, 192, (int)SZ_C2);
    // 5) q = W_q @ yn
    dmfa_gemm16<float, float, 0, false, false><<<dim3(gx, 6), 128, 0, stream>>>(
        Wqh, yn, qr, nullptr, nullptr, 96, 96, 96);
    // 6) depthwise 3x3 on q
    dmfa_dwconv<float><<<((int)SZ_C + 255) / 256, 256, 0, stream>>>(
        qr, w_q_dw, qd, 96, (int)SZ_C);
    // 7) l2 normalize q rows and k rows (over HW)
    dmfa_l2norm<<<BATCH * DIMC, 256, 0, stream>>>(qd, DIMC, DIMC);
    dmfa_l2norm<<<BATCH * DIMC, 256, 0, stream>>>(kvd, DIMC, 2 * DIMC);
    // 8) attn = softmax(temp * q k^T), 24x24 per (b,h)
    dmfa_attn<<<BATCH * NHEADS, 256, 0, stream>>>(qd, kvd, temp, attn);
    // 9) attnout = attn @ v
    dmfa_av<<<dim3(HWN / 256, BATCH * DIMC), 256, 0, stream>>>(attn, kvd, attnout);
    // 10) out1 = x + y + W_proj @ attnout
    dmfa_gemm16<float, float, 1, false, false><<<dim3(gx, 6), 128, 0, stream>>>(
        Wph, attnout, out1, x, y, 96, 96, 96);
    // 11) o1n = LN(out1, w_norm2)
    dmfa_ln<<<NPIX / 256, 256, 0, stream>>>(out1, w_norm2, o1n);
    // 12) t = W_fi @ o1n  (510 ch, store f16, masked stores on M edge)
    dmfa_gemm16<float, _Float16, 0, false, true><<<dim3(gx, 32), 128, 0, stream>>>(
        Wfih, o1n, traw, nullptr, nullptr, 510, 96, 96);
    // 13) depthwise 3x3 on t (f16 IO, f32 math)
    {
        int total = (int)((size_t)BATCH * HID2 * HWN);
        dmfa_dwconv<_Float16><<<(total + 255) / 256, 256, 0, stream>>>(
            traw, w_fdw, td, HID2, total);
    }
    // 14) g = gelu(t1) * t2
    {
        int total = (int)((size_t)BATCH * HID * HWN);
        dmfa_gelugate<<<(total + 255) / 256, 256, 0, stream>>>(td, g, total);
    }
    // 15) out = out1 + W_fo @ g   (K=255 padded to 256, masked B loads)
    dmfa_gemm16<_Float16, float, 2, true, false><<<dim3(gx, 6), 128, 0, stream>>>(
        Wfoh, g, out, out1, nullptr, 96, 255, 256);

    (void)in_sizes; (void)n_in; (void)out_size; (void)ws_size;
}
